// HiveGNN_13709535609416
// MI455X (gfx1250) — matrix operand order repaired and checked by reading, not compiled, not run on hardware
//
#include <hip/hip_runtime.h>
#include <math.h>

#define NNODES 50000
#define NEDGES 300000
#define NB     256
#define HD     128
#define NFD    32
#define EFD    9
#define GFD    16
#define NLAYER 6
#define PCH    32
#define BSIZ   13
#define NPOS   (BSIZ * BSIZ)          // 169
#define NACT   29407
#define NT_P2  1838                   // ceil(29407/16)

typedef __attribute__((ext_vector_type(16))) __bf16 v16bf;
typedef __attribute__((ext_vector_type(8)))  float  v8f;

union FragU { v16bf v; uint4 q[2]; __bf16 e[16]; };

__device__ __forceinline__ v8f wmma_bf16(v16bf a, v16bf b, v8f c) {
  return __builtin_amdgcn_wmma_f32_16x16x32_bf16(false, a, false, b, (short)0, c, false, false);
}

__device__ __forceinline__ v16bf zero_frag() {
  FragU f; f.q[0] = make_uint4(0, 0, 0, 0); f.q[1] = make_uint4(0, 0, 0, 0);
  return f.v;
}

// B fragment: packed as [frag][lane][elem], 512 bf16 per fragment
__device__ __forceinline__ v16bf load_bfrag(const __bf16* wp, int frag, int lane) {
  const uint4* p = (const uint4*)(wp + (size_t)frag * 512 + lane * 16);
  FragU f; f.q[0] = p[0]; f.q[1] = p[1];
  return f.v;
}

// A fragment from a row-major bf16 row pointer (16-bit A 16x32 layout, ISA 7.12.2)
__device__ __forceinline__ v16bf load_afrag_bf16(const __bf16* rowptr, int colbase, int lane) {
  int hi = (lane >> 4) & 1;
  FragU f;
  f.q[0] = *(const uint4*)(rowptr + colbase + hi * 8);        // K = base+hi*8 .. +7
  f.q[1] = *(const uint4*)(rowptr + colbase + 16 + hi * 8);   // K = base+16+hi*8 ..
  return f.v;
}

__device__ __forceinline__ v16bf load_afrag_f32(const float* rowptr, int colbase, int lane) {
  int hi = (lane >> 4) & 1;
  FragU f;
#pragma unroll
  for (int e = 0; e < 8; ++e) f.e[e]     = (__bf16)rowptr[colbase + hi * 8 + e];
#pragma unroll
  for (int e = 0; e < 8; ++e) f.e[8 + e] = (__bf16)rowptr[colbase + 16 + hi * 8 + e];
  return f.v;
}

__device__ __forceinline__ v16bf load_afrag_edgefeat(const float* rowptr, int lane) {
  int hi = (lane >> 4) & 1;
  FragU f;
#pragma unroll
  for (int e = 0; e < 16; ++e) {
    int k = (e < 8 ? e : e + 8) + hi * 8;
    f.e[e] = (k < EFD) ? (__bf16)rowptr[k] : (__bf16)0.0f;
  }
  return f.v;
}

__device__ __forceinline__ void atomicMaxF(float* addr, float val) {
  unsigned int* ua = (unsigned int*)addr;
  unsigned int old = *ua;
  while (__uint_as_float(old) < val) {
    unsigned int assumed = old;
    old = atomicCAS(ua, assumed, __float_as_uint(val));
    if (old == assumed) break;
  }
}

// ---------------- utility ----------------
__global__ void k_fill(float* p, float v, int n) {
  int i = blockIdx.x * blockDim.x + threadIdx.x;
  if (i < n) p[i] = v;
}

__global__ void k_f32_to_bf16(const float* __restrict__ in, __bf16* __restrict__ out, int n) {
  int i = blockIdx.x * blockDim.x + threadIdx.x;
  if (i < n) out[i] = (__bf16)in[i];
}

// ---------------- weight packing: row-major (K,N) f32 -> bf16 B-fragments ----------------
__global__ void k_pack(const float* __restrict__ src, __bf16* __restrict__ dst,
                       int Korig, int Norig, int KC, int NT) {
  int idx = blockIdx.x * blockDim.x + threadIdx.x;
  int total = KC * NT * 512;
  if (idx >= total) return;
  int e = idx & 15, lane = (idx >> 4) & 31, f = idx >> 9;
  int nt = f % NT, kc = f / NT;
  int k = kc * 32 + (e < 8 ? e : e + 8) + ((lane >= 16) ? 8 : 0);  // B layout: lane = N col, elem = K
  int n = nt * 16 + (lane & 15);
  float v = (k < Korig && n < Norig) ? src[(size_t)k * Norig + n] : 0.0f;
  dst[idx] = (__bf16)v;
}

// conv1 weights (oc,ic,3,3) -> B-fragments with K = tap*128+ic, N = oc  (36 chunks x 2 tiles)
__global__ void k_packc1(const float* __restrict__ c1w, __bf16* __restrict__ dst) {
  int idx = blockIdx.x * blockDim.x + threadIdx.x;
  if (idx >= 36 * 2 * 512) return;
  int e = idx & 15, lane = (idx >> 4) & 31, f = idx >> 9;
  int nt = f & 1, kc = f >> 1;
  int k = kc * 32 + (e < 8 ? e : e + 8) + ((lane >= 16) ? 8 : 0);   // 0..1151
  int oc = nt * 16 + (lane & 15);                  // 0..31
  int tap = k / HD, ic = k % HD;
  float v = c1w[(((size_t)oc * HD + ic) * 3 + tap / 3) * 3 + (tap % 3)];
  dst[idx] = (__bf16)v;
}

// ---------------- node projection: h = relu(nf @ Wp + bp) ----------------
__global__ void k_proj(const float* __restrict__ nodef, const __bf16* __restrict__ wp,
                       const float* __restrict__ bias, float* __restrict__ h32,
                       __bf16* __restrict__ h16) {
  int wave = (blockIdx.x * blockDim.x + threadIdx.x) >> 5;
  int lane = threadIdx.x & 31;
  if (wave >= NNODES / 16) return;
  int row0 = wave * 16;
  int hi = (lane >> 4) & 1;
  int r = row0 + (lane & 15);
  v16bf a = load_afrag_f32(nodef + (size_t)r * NFD, 0, lane);
#pragma unroll
  for (int nt = 0; nt < 8; ++nt) {
    v8f acc{};
    acc = wmma_bf16(a, load_bfrag(wp, nt, lane), acc);
    int col = nt * 16 + (lane & 15);
    float bv = bias[col];
#pragma unroll
    for (int q = 0; q < 8; ++q) {
      float x = acc[q] + bv; x = x > 0.f ? x : 0.f;
      int rr = row0 + q + hi * 8;
      h32[(size_t)rr * HD + col] = x;
      h16[(size_t)rr * HD + col] = (__bf16)x;
    }
  }
}

// ---------------- edge message MLP + scatter-add (weights staged in LDS) ----------------
#define EW 8
#define W1_FRAGS (9 * 8)
#define W2_FRAGS (4 * 8)
#define W1_Q ((W1_FRAGS * 512 * 2) / 16)   // 4608 uint4
#define W2_Q ((W2_FRAGS * 512 * 2) / 16)   // 2048 uint4
__global__ void k_edge(const __bf16* __restrict__ h16, const float* __restrict__ ef,
                       const int* __restrict__ esrc, const int* __restrict__ edst,
                       const __bf16* __restrict__ w1p, const float* __restrict__ b1,
                       const __bf16* __restrict__ w2p, const float* __restrict__ b2,
                       float* __restrict__ agg) {
  __shared__ __bf16 hid[EW][16 * HD];   // 32 KB, wave-private tiles
  extern __shared__ char dynlds[];      // 104 KB packed weights
  __bf16* w1s = (__bf16*)dynlds;
  __bf16* w2s = w1s + W1_FRAGS * 512;
  {
    uint4* s = (uint4*)dynlds;
    const uint4* g1 = (const uint4*)w1p;
    const uint4* g2 = (const uint4*)w2p;
    for (int i = threadIdx.x; i < W1_Q + W2_Q; i += blockDim.x)
      s[i] = (i < W1_Q) ? g1[i] : g2[i - W1_Q];
  }
  __syncthreads();

  int wid  = threadIdx.x >> 5;
  int wave = (blockIdx.x * blockDim.x + threadIdx.x) >> 5;
  int lane = threadIdx.x & 31;
  if (wave >= NEDGES / 16) return;
  int hi = (lane >> 4) & 1;
  int erow = wave * 16 + (lane & 15);
  int nsrc = esrc[erow], ndst = edst[erow];

  v8f acc[8];
#pragma unroll
  for (int t = 0; t < 8; ++t) acc[t] = v8f{};
  for (int kc = 0; kc < 9; ++kc) {
    v16bf a;
    if (kc < 4)      a = load_afrag_bf16(h16 + (size_t)ndst * HD, kc * 32, lane);
    else if (kc < 8) a = load_afrag_bf16(h16 + (size_t)nsrc * HD, (kc - 4) * 32, lane);
    else             a = load_afrag_edgefeat(ef + (size_t)erow * EFD, lane);
#pragma unroll
    for (int nt = 0; nt < 8; ++nt)
      acc[nt] = wmma_bf16(a, load_bfrag(w1s, kc * 8 + nt, lane), acc[nt]);
  }
#pragma unroll
  for (int nt = 0; nt < 8; ++nt) {
    int col = nt * 16 + (lane & 15);
    float bv = b1[col];
#pragma unroll
    for (int q = 0; q < 8; ++q) {
      float x = acc[nt][q] + bv; x = x > 0.f ? x : 0.f;
      hid[wid][(q + hi * 8) * HD + col] = (__bf16)x;
    }
  }
  v8f acc2[8];
#pragma unroll
  for (int t = 0; t < 8; ++t) acc2[t] = v8f{};
  const __bf16* hrow = &hid[wid][(lane & 15) * HD];
  for (int kc = 0; kc < 4; ++kc) {
    v16bf a = load_afrag_bf16(hrow, kc * 32, lane);
#pragma unroll
    for (int nt = 0; nt < 8; ++nt)
      acc2[nt] = wmma_bf16(a, load_bfrag(w2s, kc * 8 + nt, lane), acc2[nt]);
  }
  int nd[8];
#pragma unroll
  for (int q = 0; q < 8; ++q) nd[q] = edst[wave * 16 + q + hi * 8];
#pragma unroll
  for (int nt = 0; nt < 8; ++nt) {
    int col = nt * 16 + (lane & 15);
    float bv = b2[col];
#pragma unroll
    for (int q = 0; q < 8; ++q)
      atomicAdd(&agg[(size_t)nd[q] * HD + col], acc2[nt][q] + bv);
  }
}

// ---------------- node update MLP + residual + LayerNorm (weights staged in LDS) ----------------
#define UW 4
#define U1_FRAGS (8 * 8)
#define U2_FRAGS (4 * 8)
#define U1_Q ((U1_FRAGS * 512 * 2) / 16)   // 4096 uint4
#define U2_Q ((U2_FRAGS * 512 * 2) / 16)   // 2048 uint4
__global__ void k_update(const float* __restrict__ agg, const __bf16* __restrict__ w1p,
                         const float* __restrict__ b1, const __bf16* __restrict__ w2p,
                         const float* __restrict__ b2, const float* __restrict__ lng,
                         const float* __restrict__ lnb, float* __restrict__ h32,
                         __bf16* __restrict__ h16) {
  __shared__ __bf16 hid[UW][16 * HD];
  __shared__ float  xbuf[UW][16 * HD];
  __shared__ float  mu[UW][16], rs[UW][16];
  extern __shared__ char dynlds[];
  __bf16* w1s = (__bf16*)dynlds;
  __bf16* w2s = w1s + U1_FRAGS * 512;
  {
    uint4* s = (uint4*)dynlds;
    const uint4* g1 = (const uint4*)w1p;
    const uint4* g2 = (const uint4*)w2p;
    for (int i = threadIdx.x; i < U1_Q + U2_Q; i += blockDim.x)
      s[i] = (i < U1_Q) ? g1[i] : g2[i - U1_Q];
  }
  __syncthreads();

  int wid  = threadIdx.x >> 5;
  int wave = (blockIdx.x * blockDim.x + threadIdx.x) >> 5;
  int lane = threadIdx.x & 31;
  if (wave >= NNODES / 16) return;
  int hi = (lane >> 4) & 1;
  int row0 = wave * 16;
  int r = row0 + (lane & 15);

  v8f acc[8];
#pragma unroll
  for (int t = 0; t < 8; ++t) acc[t] = v8f{};
  for (int kc = 0; kc < 8; ++kc) {
    v16bf a = (kc < 4) ? load_afrag_bf16(h16 + (size_t)r * HD, kc * 32, lane)
                       : load_afrag_f32(agg + (size_t)r * HD, (kc - 4) * 32, lane);
#pragma unroll
    for (int nt = 0; nt < 8; ++nt)
      acc[nt] = wmma_bf16(a, load_bfrag(w1s, kc * 8 + nt, lane), acc[nt]);
  }
#pragma unroll
  for (int nt = 0; nt < 8; ++nt) {
    int col = nt * 16 + (lane & 15);
    float bv = b1[col];
#pragma unroll
    for (int q = 0; q < 8; ++q) {
      float x = acc[nt][q] + bv; x = x > 0.f ? x : 0.f;
      hid[wid][(q + hi * 8) * HD + col] = (__bf16)x;
    }
  }
  v8f acc2[8];
#pragma unroll
  for (int t = 0; t < 8; ++t) acc2[t] = v8f{};
  const __bf16* hrow = &hid[wid][(lane & 15) * HD];
  for (int kc = 0; kc < 4; ++kc) {
    v16bf a = load_afrag_bf16(hrow, kc * 32, lane);
#pragma unroll
    for (int nt = 0; nt < 8; ++nt)
      acc2[nt] = wmma_bf16(a, load_bfrag(w2s, kc * 8 + nt, lane), acc2[nt]);
  }
#pragma unroll
  for (int nt = 0; nt < 8; ++nt) {
    int col = nt * 16 + (lane & 15);
    float bv = b2[col];
#pragma unroll
    for (int q = 0; q < 8; ++q) {
      int rr = q + hi * 8;
      xbuf[wid][rr * HD + col] = acc2[nt][q] + bv + h32[(size_t)(row0 + rr) * HD + col];
    }
  }
  if (lane < 16) {
    float s = 0.f, s2 = 0.f;
    for (int c = 0; c < HD; ++c) { float x = xbuf[wid][lane * HD + c]; s += x; s2 += x * x; }
    float m = s * (1.0f / HD);
    mu[wid][lane] = m;
    rs[wid][lane] = rsqrtf(s2 * (1.0f / HD) - m * m + 1e-5f);
  }
#pragma unroll
  for (int i = 0; i < 64; ++i) {
    int idx = lane * 64 + i;
    int row = idx >> 7, col = idx & (HD - 1);
    float x = (xbuf[wid][idx] - mu[wid][row]) * rs[wid][row] * lng[col] + lnb[col];
    size_t o = (size_t)(row0 + row) * HD + col;
    h32[o] = x; h16[o] = (__bf16)x;
  }
}

// ---------------- spatial grid scatter (NHWC layout) ----------------
__global__ void k_scatter(const float* __restrict__ h32, const int* __restrict__ batch,
                          const int* __restrict__ pos, float* __restrict__ grid) {
  int idx = blockIdx.x * blockDim.x + threadIdx.x;
  if (idx >= NNODES * HD) return;
  int n = idx >> 7, k = idx & (HD - 1);
  int b = batch[n], rr = pos[n * 2], cc = pos[n * 2 + 1];
  atomicAdd(&grid[((size_t)b * NPOS + rr * BSIZ + cc) * HD + k], h32[idx]);
}

// ---------------- conv1 3x3 SAME via WMMA implicit GEMM + BN + relu ----------------
#define CW 8
#define C1_FRAGS (36 * 2)
#define C1_Q ((C1_FRAGS * 512 * 2) / 16)   // 4608 uint4
__global__ void k_conv1w(const __bf16* __restrict__ gridb, const __bf16* __restrict__ wp,
                         const float* __restrict__ g, const float* __restrict__ bb,
                         const float* __restrict__ mm, const float* __restrict__ vvar,
                         float* __restrict__ p1) {
  extern __shared__ char dynlds[];   // 72 KB packed conv weights
  __bf16* ws = (__bf16*)dynlds;
  {
    uint4* s = (uint4*)dynlds;
    const uint4* gsrc = (const uint4*)wp;
    for (int i = threadIdx.x; i < C1_Q; i += blockDim.x) s[i] = gsrc[i];
  }
  __syncthreads();

  int wave = (blockIdx.x * blockDim.x + threadIdx.x) >> 5;
  int lane = threadIdx.x & 31;
  if (wave >= (NB * NPOS) / 16) return;   // 2704 tiles
  int hi = (lane >> 4) & 1;
  int rr = wave * 16 + (lane & 15);
  int b = rr / NPOS, p = rr % NPOS;
  int r = p / BSIZ, c = p % BSIZ;

  v8f acc[2]; acc[0] = v8f{}; acc[1] = v8f{};
  for (int tap = 0; tap < 9; ++tap) {
    int r2 = r + tap / 3 - 1, c2 = c + tap % 3 - 1;
    bool valid = (r2 >= 0) && (r2 < BSIZ) && (c2 >= 0) && (c2 < BSIZ);
    const __bf16* rowptr = gridb + ((size_t)b * NPOS + (valid ? r2 * BSIZ + c2 : 0)) * HD;
#pragma unroll
    for (int sub = 0; sub < 4; ++sub) {
      v16bf a = valid ? load_afrag_bf16(rowptr, sub * 32, lane) : zero_frag();
#pragma unroll
      for (int nt = 0; nt < 2; ++nt)
        acc[nt] = wmma_bf16(a, load_bfrag(ws, (tap * 4 + sub) * 2 + nt, lane), acc[nt]);
    }
  }
#pragma unroll
  for (int nt = 0; nt < 2; ++nt) {
    int oc = nt * 16 + (lane & 15);
    float sc = g[oc] * rsqrtf(vvar[oc] + 1e-5f);
    float sh = bb[oc] - mm[oc] * sc;
#pragma unroll
    for (int q = 0; q < 8; ++q) {
      int rrow = wave * 16 + q + hi * 8;
      int b2 = rrow / NPOS, pp = rrow % NPOS;
      float x = acc[nt][q] * sc + sh; x = x > 0.f ? x : 0.f;
      p1[((size_t)b2 * PCH + oc) * NPOS + pp] = x;
    }
  }
}

// ---------------- conv2 1x1 + BN + relu + concat globals -> pvec (B,354) ----------------
__global__ void k_conv2(const float* __restrict__ p1, const float* __restrict__ w,
                        const float* __restrict__ g, const float* __restrict__ bb,
                        const float* __restrict__ mm, const float* __restrict__ vvar,
                        const float* __restrict__ gf, float* __restrict__ pvec) {
  int idx = blockIdx.x * blockDim.x + threadIdx.x;
  if (idx >= NB * 354) return;
  int b = idx / 354, j = idx % 354;
  if (j < 2 * NPOS) {
    int oc = j / NPOS, p = j % NPOS;
    float s = 0.f;
    for (int ic = 0; ic < PCH; ++ic)
      s += p1[((size_t)b * PCH + ic) * NPOS + p] * w[oc * PCH + ic];
    float sc = g[oc] * rsqrtf(vvar[oc] + 1e-5f);
    s = s * sc + (bb[oc] - mm[oc] * sc);
    pvec[idx] = s > 0.f ? s : 0.f;
  } else {
    pvec[idx] = gf[b * GFD + (j - 2 * NPOS)];
  }
}

// ---------------- generic FC + relu (small) ----------------
__global__ void k_fc_relu(const float* __restrict__ in, const float* __restrict__ w,
                          const float* __restrict__ bias, float* __restrict__ out,
                          int M, int K, int N) {
  int idx = blockIdx.x * blockDim.x + threadIdx.x;
  if (idx >= M * N) return;
  int row = idx / N, col = idx % N;
  float s = bias[col];
  for (int k = 0; k < K; ++k) s += in[(size_t)row * K + k] * w[(size_t)k * N + col];
  out[idx] = s > 0.f ? s : 0.f;
}

// ---------------- policy FC2 via WMMA, block-staged weights ----------------
// 512 threads = 16 waves; block = one group of 8 N-tiles, wave w = M-tile w.
__global__ void k_pfc2(const __bf16* __restrict__ pf1b, const __bf16* __restrict__ wp,
                       const float* __restrict__ bias, float* __restrict__ logits) {
  extern __shared__ char dynlds[];       // 64 fragments = 64 KB
  __bf16* ws = (__bf16*)dynlds;
  int ntg = blockIdx.x;
  {
    uint4* s = (uint4*)dynlds;
    for (int i = threadIdx.x; i < 64 * 64; i += blockDim.x) {   // 64 uint4 per fragment
      int f = i >> 6, q = i & 63;
      int kc = f >> 3, t = f & 7;
      int nt = ntg * 8 + t;
      uint4 val = make_uint4(0, 0, 0, 0);
      if (nt < NT_P2)
        val = ((const uint4*)(wp + (size_t)(kc * NT_P2 + nt) * 512))[q];
      s[i] = val;
    }
  }
  __syncthreads();

  int mt = threadIdx.x >> 5;          // 0..15
  int lane = threadIdx.x & 31;
  int hi = (lane >> 4) & 1;
  int r = mt * 16 + (lane & 15);
  v16bf a[8];
#pragma unroll
  for (int kc = 0; kc < 8; ++kc) a[kc] = load_afrag_bf16(pf1b + (size_t)r * 256, kc * 32, lane);
  for (int t = 0; t < 8; ++t) {
    int nt = ntg * 8 + t;
    v8f acc{};
#pragma unroll
    for (int kc = 0; kc < 8; ++kc)
      acc = wmma_bf16(a[kc], load_bfrag(ws, kc * 8 + t, lane), acc);
    int col = nt * 16 + (lane & 15);
    if (col < NACT) {
      float bv = bias[col];
#pragma unroll
      for (int q = 0; q < 8; ++q)
        logits[(size_t)(mt * 16 + q + hi * 8) * NACT + col] = acc[q] + bv;
    }
  }
}

// ---------------- value-head pooling ----------------
__global__ void k_pool(const float* __restrict__ h32, const int* __restrict__ batch,
                       float* meanp, float* maxp, float* cnt) {
  int idx = blockIdx.x * blockDim.x + threadIdx.x;
  if (idx >= NNODES * HD) return;
  int n = idx >> 7, k = idx & (HD - 1);
  int b = batch[n];
  float x = h32[idx];
  atomicAdd(&meanp[(size_t)b * HD + k], x);
  atomicMaxF(&maxp[(size_t)b * HD + k], x);
  if (k == 0) atomicAdd(&cnt[b], 1.0f);
}

__global__ void k_comb(const float* __restrict__ meanp, const float* __restrict__ maxp,
                       const float* __restrict__ cnt, const float* __restrict__ gf,
                       float* __restrict__ comb) {
  int idx = blockIdx.x * blockDim.x + threadIdx.x;
  if (idx >= NB * (2 * HD + GFD)) return;
  int b = idx / (2 * HD + GFD), j = idx % (2 * HD + GFD);
  float v;
  if (j < HD) {
    float c = cnt[b]; c = c > 1.0f ? c : 1.0f;
    v = meanp[(size_t)b * HD + j] / c;
  } else if (j < 2 * HD) {
    float mx = maxp[(size_t)b * HD + (j - HD)];
    v = isinf(mx) ? 0.0f : mx;
  } else {
    v = gf[b * GFD + (j - 2 * HD)];
  }
  comb[idx] = v;
}

__global__ void k_value(const float* __restrict__ vf1, const float* __restrict__ w,
                        const float* __restrict__ b, float* __restrict__ out) {
  int i = blockIdx.x * blockDim.x + threadIdx.x;
  if (i >= NB) return;
  float s = b[0];
  for (int k = 0; k < 256; ++k) s += vf1[(size_t)i * 256 + k] * w[k];
  out[i] = tanhf(s);
}

// =====================================================================
extern "C" void kernel_launch(void* const* d_in, const int* in_sizes, int n_in,
                              void* d_out, int out_size, void* d_ws, size_t ws_size,
                              hipStream_t stream) {
  const float* nodef = (const float*)d_in[0];
  const float* edgef = (const float*)d_in[1];
  const float* gfeat = (const float*)d_in[2];
  const int*   eidx  = (const int*)d_in[3];
  const int*   batch = (const int*)d_in[4];
  const int*   pos   = (const int*)d_in[5];
  const float* projw = (const float*)d_in[6];  const float* projb = (const float*)d_in[7];
  const float* mw1 = (const float*)d_in[8];    const float* mb1 = (const float*)d_in[9];
  const float* mw2 = (const float*)d_in[10];   const float* mb2 = (const float*)d_in[11];
  const float* uw1 = (const float*)d_in[12];   const float* ub1 = (const float*)d_in[13];
  const float* uw2 = (const float*)d_in[14];   const float* ub2 = (const float*)d_in[15];
  const float* lng = (const float*)d_in[16];   const float* lnb = (const float*)d_in[17];
  const float* c1w = (const float*)d_in[18];
  const float* bn1g = (const float*)d_in[19];  const float* bn1b = (const float*)d_in[20];
  const float* bn1m = (const float*)d_in[21];  const float* bn1v = (const float*)d_in[22];
  const float* c2w = (const float*)d_in[23];
  const float* bn2g = (const float*)d_in[24];  const float* bn2b = (const float*)d_in[25];
  const float* bn2m = (const float*)d_in[26];  const float* bn2v = (const float*)d_in[27];
  const float* p1w = (const float*)d_in[28];   const float* p1b = (const float*)d_in[29];
  const float* p2w = (const float*)d_in[30];   const float* p2b = (const float*)d_in[31];
  const float* v1w = (const float*)d_in[32];   const float* v1b = (const float*)d_in[33];
  const float* v2w = (const float*)d_in[34];   const float* v2b = (const float*)d_in[35];

  const int* esrc = eidx;
  const int* edst = eidx + NEDGES;

  // workspace carve
  char* base = (char*)d_ws; size_t off = 0;
  auto carve = [&](size_t bytes) { void* p = base + off; off += (bytes + 255) & ~(size_t)255; return p; };
  float*  h32   = (float*) carve((size_t)NNODES * HD * 4);
  __bf16* h16   = (__bf16*)carve((size_t)NNODES * HD * 2);
  float*  agg   = (float*) carve((size_t)NNODES * HD * 4);
  __bf16* wpP   = (__bf16*)carve((size_t)1 * 8 * 512 * 2);
  __bf16* wpM1  = (__bf16*)carve((size_t)NLAYER * 9 * 8 * 512 * 2);
  __bf16* wpM2  = (__bf16*)carve((size_t)NLAYER * 4 * 8 * 512 * 2);
  __bf16* wpU1  = (__bf16*)carve((size_t)NLAYER * 8 * 8 * 512 * 2);
  __bf16* wpU2  = (__bf16*)carve((size_t)NLAYER * 4 * 8 * 512 * 2);
  __bf16* wpP2  = (__bf16*)carve((size_t)8 * NT_P2 * 512 * 2);
  __bf16* wpC1  = (__bf16*)carve((size_t)C1_FRAGS * 512 * 2);
  float*  grid  = (float*) carve((size_t)NB * NPOS * HD * 4);    // NHWC
  __bf16* gridb = (__bf16*)carve((size_t)NB * NPOS * HD * 2);    // NHWC bf16
  float*  p1    = (float*) carve((size_t)NB * PCH * NPOS * 4);
  float*  pvec  = (float*) carve((size_t)NB * 354 * 4);
  float*  pf1   = (float*) carve((size_t)NB * 256 * 4);
  __bf16* pf1b  = (__bf16*)carve((size_t)NB * 256 * 2);
  float*  meanp = (float*) carve((size_t)NB * HD * 4);
  float*  maxp  = (float*) carve((size_t)NB * HD * 4);
  float*  cnt   = (float*) carve((size_t)NB * 4);
  float*  comb  = (float*) carve((size_t)NB * (2 * HD + GFD) * 4);
  float*  vf1   = (float*) carve((size_t)NB * 256 * 4);

  float* logits = (float*)d_out;
  float* value  = (float*)d_out + (size_t)NB * NACT;

  auto grid1d = [](int n, int bs) { return dim3((unsigned)((n + bs - 1) / bs)); };

  // ---- pack weights into bf16 WMMA B-fragments ----
  k_pack<<<grid1d(1 * 8 * 512, 256), 256, 0, stream>>>(projw, wpP, NFD, HD, 1, 8);
  for (int l = 0; l < NLAYER; ++l) {
    k_pack<<<grid1d(9 * 8 * 512, 256), 256, 0, stream>>>(mw1 + (size_t)l * 265 * HD, wpM1 + (size_t)l * 9 * 8 * 512, 265, HD, 9, 8);
    k_pack<<<grid1d(4 * 8 * 512, 256), 256, 0, stream>>>(mw2 + (size_t)l * HD * HD,  wpM2 + (size_t)l * 4 * 8 * 512, HD, HD, 4, 8);
    k_pack<<<grid1d(8 * 8 * 512, 256), 256, 0, stream>>>(uw1 + (size_t)l * 256 * HD, wpU1 + (size_t)l * 8 * 8 * 512, 256, HD, 8, 8);
    k_pack<<<grid1d(4 * 8 * 512, 256), 256, 0, stream>>>(uw2 + (size_t)l * HD * HD,  wpU2 + (size_t)l * 4 * 8 * 512, HD, HD, 4, 8);
  }
  k_pack<<<grid1d(8 * NT_P2 * 512, 256), 256, 0, stream>>>(p2w, wpP2, 256, NACT, 8, NT_P2);
  k_packc1<<<grid1d(C1_FRAGS * 512, 256), 256, 0, stream>>>(c1w, wpC1);

  // ---- node projection ----
  k_proj<<<grid1d((NNODES / 16) * 32, 256), 256, 0, stream>>>(nodef, wpP, projb, h32, h16);

  // ---- message-passing layers ----
  const int edgeShm = (W1_Q + W2_Q) * 16;
  const int updShm  = (U1_Q + U2_Q) * 16;
  for (int l = 0; l < NLAYER; ++l) {
    k_fill<<<grid1d(NNODES * HD, 256), 256, 0, stream>>>(agg, 0.0f, NNODES * HD);
    k_edge<<<grid1d((NEDGES / 16) * 32, EW * 32), EW * 32, edgeShm, stream>>>(
        h16, edgef, esrc, edst,
        wpM1 + (size_t)l * 9 * 8 * 512, mb1 + (size_t)l * HD,
        wpM2 + (size_t)l * 4 * 8 * 512, mb2 + (size_t)l * HD, agg);
    k_update<<<grid1d((NNODES / 16) * 32, UW * 32), UW * 32, updShm, stream>>>(
        agg, wpU1 + (size_t)l * 8 * 8 * 512, ub1 + (size_t)l * HD,
        wpU2 + (size_t)l * 4 * 8 * 512, ub2 + (size_t)l * HD,
        lng + (size_t)l * HD, lnb + (size_t)l * HD, h32, h16);
  }

  // ---- policy head ----
  k_fill<<<grid1d(NB * NPOS * HD, 256), 256, 0, stream>>>(grid, 0.0f, NB * NPOS * HD);
  k_scatter<<<grid1d(NNODES * HD, 256), 256, 0, stream>>>(h32, batch, pos, grid);
  k_f32_to_bf16<<<grid1d(NB * NPOS * HD, 256), 256, 0, stream>>>(grid, gridb, NB * NPOS * HD);
  k_conv1w<<<grid1d(((NB * NPOS) / 16) * 32, CW * 32), CW * 32, C1_Q * 16, stream>>>(
      gridb, wpC1, bn1g, bn1b, bn1m, bn1v, p1);
  k_conv2<<<grid1d(NB * 354, 256), 256, 0, stream>>>(p1, c2w, bn2g, bn2b, bn2m, bn2v, gfeat, pvec);
  k_fc_relu<<<grid1d(NB * 256, 256), 256, 0, stream>>>(pvec, p1w, p1b, pf1, NB, 354, 256);
  k_f32_to_bf16<<<grid1d(NB * 256, 256), 256, 0, stream>>>(pf1, pf1b, NB * 256);
  k_pfc2<<<dim3((NT_P2 + 7) / 8), 512, 64 * 64 * 16, stream>>>(pf1b, wpP2, p2b, logits);

  // ---- value head ----
  k_fill<<<grid1d(NB * HD, 256), 256, 0, stream>>>(meanp, 0.0f, NB * HD);
  k_fill<<<grid1d(NB * HD, 256), 256, 0, stream>>>(maxp, -INFINITY, NB * HD);
  k_fill<<<grid1d(NB, 256), 256, 0, stream>>>(cnt, 0.0f, NB);
  k_pool<<<grid1d(NNODES * HD, 256), 256, 0, stream>>>(h32, batch, meanp, maxp, cnt);
  k_comb<<<grid1d(NB * (2 * HD + GFD), 256), 256, 0, stream>>>(meanp, maxp, cnt, gfeat, comb);
  k_fc_relu<<<grid1d(NB * 256, 256), 256, 0, stream>>>(comb, v1w, v1b, vf1, NB, 2 * HD + GFD, 256);
  k_value<<<grid1d(NB, 256), 256, 0, stream>>>(vf1, v2w, v2b, value);
}
